// CausalSelfAttention_55551107007172
// MI455X (gfx1250) — matrix operand order repaired and checked by reading, not compiled, not run on hardware
//
#include <hip/hip_runtime.h>
#include <hip/hip_bf16.h>
#include <math.h>

// ---------------------------------------------------------------------------
// Causal self-attention for MI455X (gfx1250, wave32, WMMA, TDM).
// Pipeline: f32->bf16 convert, W transposes, QKV GEMM (TDM-staged bf16 WMMA,
// f32 acc), flash-attention (bf16 WMMA + online softmax), out-proj GEMM.
// ---------------------------------------------------------------------------

#define B_SZ 2
#define T_SZ 2048
#define C_SZ 1024
#define H_SZ 16
#define D_SZ 64

typedef __attribute__((ext_vector_type(16))) __bf16 v16bf;
typedef __attribute__((ext_vector_type(8)))  __bf16 v8bf;
typedef __attribute__((ext_vector_type(8)))  float  v8f;
typedef __attribute__((ext_vector_type(4)))  unsigned int v4u;
typedef __attribute__((ext_vector_type(4)))  int    v4i;
typedef __attribute__((ext_vector_type(8)))  int    v8i;
typedef __attribute__((ext_vector_type(4)))  float  f32x4;

union BF16Frag { v16bf v; v8bf h[2]; };
union FAcc     { v8f   v; float f[8]; };

__device__ __forceinline__ __bf16 f2bf(float f) {
  unsigned u = __float_as_uint(f);
  unsigned r = u + 0x7FFFu + ((u >> 16) & 1u);   // round-to-nearest-even
  unsigned short s = (unsigned short)(r >> 16);
  __bf16 b;
  __builtin_memcpy(&b, &s, 2);
  return b;
}

// LDS byte offset of a __shared__ object (addrspace(3) pointer value).
typedef __attribute__((address_space(3))) void lds_void;
__device__ __forceinline__ unsigned lds_off(void* p) {
  return (unsigned)(unsigned long long)(lds_void*)p;
}

// ---------------------------------------------------------------------------
// TDM: DMA a 2-D tile (rows x row_elems of bf16) from global into LDS,
// inserting a 4-DWORD pad after every row (row_elems*2B must be 32 DWORDs).
// D# layout per CDNA5 ISA ch.8 (group0: addr/type, group1: dims/strides/pad).
// ---------------------------------------------------------------------------
__device__ __forceinline__ void tdm_load_tile_bf16(
    const __bf16* gsrc, unsigned lds_addr, unsigned rows, unsigned row_elems,
    unsigned src_stride_elems) {
  const unsigned long long ga = (unsigned long long)(const void*)gsrc;
  v4u g0;
  g0.x = 1u;                                         // count=1 (valid), user D#
  g0.y = lds_addr;                                   // lds_addr[31:0]
  g0.z = (unsigned)ga;                               // global_addr[31:0]
  g0.w = (unsigned)((ga >> 32) & 0x01FFFFFFu)        // global_addr[56:32]
         | (2u << 30);                               // type=2 ("image")
  const unsigned tdim0 = src_stride_elems;           // >= tile extent (in-bounds)
  const unsigned tdim1 = rows;
  v8i g1;
  g1[0] = (int)((1u << 16)                           // data_size: 2 bytes
                | (1u << 20)                         // pad_enable
                | (4u << 22)                         // pad_interval: 32 DWORDs
                | (3u << 25));                       // pad_amount:   4 DWORDs
  g1[1] = (int)((tdim0 & 0xFFFFu) << 16);            // tensor_dim0[15:0]
  g1[2] = (int)(((tdim0 >> 16) & 0xFFFFu) | ((tdim1 & 0xFFFFu) << 16));
  g1[3] = (int)(((tdim1 >> 16) & 0xFFFFu) | ((row_elems & 0xFFFFu) << 16)); // tile_dim0
  g1[4] = (int)(rows & 0xFFFFu);                     // tile_dim1 (tile_dim2=0)
  g1[5] = (int)src_stride_elems;                     // tensor_dim0_stride[31:0]
  g1[6] = 0;                                         // stride[47:32] | dim1_stride lo
  g1[7] = 0;
  v4i g2 = {0, 0, 0, 0};
  v4i g3 = {0, 0, 0, 0};
#if defined(__clang_major__) && __clang_major__ >= 23
  v8i g4 = {0, 0, 0, 0, 0, 0, 0, 0};
  __builtin_amdgcn_tensor_load_to_lds(g0, g1, g2, g3, g4, 0);
#else
  __builtin_amdgcn_tensor_load_to_lds(g0, g1, g2, g3, 0);
#endif
}

// ---------------------------------------------------------------------------
// 1) elementwise fp32 -> bf16 (4 elements / thread, vectorized read)
// ---------------------------------------------------------------------------
__global__ __launch_bounds__(256) void cvt_f32_bf16(const float* __restrict__ in,
                                                    __bf16* __restrict__ out, int n) {
  int i = (blockIdx.x * blockDim.x + threadIdx.x) * 4;
  if (i + 3 < n) {
    f32x4 v = *(const f32x4*)(in + i);
    out[i + 0] = f2bf(v.x);
    out[i + 1] = f2bf(v.y);
    out[i + 2] = f2bf(v.z);
    out[i + 3] = f2bf(v.w);
  }
}

// ---------------------------------------------------------------------------
// 2) tiled transpose + convert:  out[n*K + k] = bf16(in[k*N + n])
// ---------------------------------------------------------------------------
__global__ __launch_bounds__(256) void transpose_f32_bf16(const float* __restrict__ in,
                                                          __bf16* __restrict__ out,
                                                          int K, int N) {
  __shared__ float tile[32][33];
  int n0 = blockIdx.x * 32, k0 = blockIdx.y * 32;
  int tx = threadIdx.x, ty = threadIdx.y;           // 32 x 8
#pragma unroll
  for (int i = 0; i < 32; i += 8)
    tile[ty + i][tx] = in[(size_t)(k0 + ty + i) * N + n0 + tx];
  __syncthreads();
#pragma unroll
  for (int i = 0; i < 32; i += 8)
    out[(size_t)(n0 + ty + i) * K + k0 + tx] = f2bf(tile[tx][ty + i]);
}

// ---------------------------------------------------------------------------
// 3) bf16 GEMM, f32 accumulate:  C[M,N] = A[M,K] @ Bt[N,K]^T + bias[N]
//    128x128 block tile, BK=64, tiles staged by the Tensor Data Mover with
//    TENSORcnt double buffering. 8 waves as 2(M) x 4(N); 16 WMMA / k-step.
//    LDS rows padded by TDM: 64 bf16 (32 DW) + 4 DW pad -> 72-elem stride.
//    MODE 0: fp32 output [M,N].  MODE 1: qkv scatter epilogue into
//            Q[B,H,T,D], K[B,H,T,D], Vt[B,H,D,T]  (bf16).
// ---------------------------------------------------------------------------
#define LROW 72

template <int MODE>
__global__ __launch_bounds__(256) void gemm_bf16(
    const __bf16* __restrict__ A, const __bf16* __restrict__ Bt,
    const float* __restrict__ bias, float* __restrict__ outp,
    __bf16* __restrict__ Qb, __bf16* __restrict__ Kb, __bf16* __restrict__ Vt,
    int M, int N, int K) {
  __shared__ __attribute__((aligned(16))) __bf16 As[2][128 * LROW];
  __shared__ __attribute__((aligned(16))) __bf16 Bs[2][128 * LROW];

  const int tid  = threadIdx.x;
  const int lane = tid & 31;
  const int w    = tid >> 5;            // 0..7
  const int wm   = w >> 2;              // 0..1  (64 rows each)
  const int wn   = w & 3;               // 0..3  (32 cols each)
  const int n16  = lane & 15;
  const int half = lane >> 4;

  const int n0 = blockIdx.x * 128;
  const int m0 = blockIdx.y * 128;
  const int nk = K >> 6;                // BK = 64

  FAcc acc[4][2];
#pragma unroll
  for (int mt = 0; mt < 4; ++mt)
#pragma unroll
    for (int nt = 0; nt < 2; ++nt)
#pragma unroll
      for (int r = 0; r < 8; ++r) acc[mt][nt].f[r] = 0.0f;

  // prologue: DMA k-slice 0 into buffer 0 (wave 0 drives the TDM)
  if (w == 0) {
    tdm_load_tile_bf16(A  + (size_t)m0 * K, lds_off(As[0]), 128, 64, K);
    tdm_load_tile_bf16(Bt + (size_t)n0 * K, lds_off(Bs[0]), 128, 64, K);
  }

  for (int ks = 0; ks < nk; ++ks) {
    const int cur = ks & 1;
    if (w == 0) {
      if (ks + 1 < nk) {
        // buffer cur^1 was consumed at iteration ks-1 (end-of-iter barrier
        // already crossed by every wave) -> safe to overwrite via DMA.
        const int k1 = (ks + 1) << 6;
        tdm_load_tile_bf16(A  + (size_t)m0 * K + k1, lds_off(As[cur ^ 1]), 128, 64, K);
        tdm_load_tile_bf16(Bt + (size_t)n0 * K + k1, lds_off(Bs[cur ^ 1]), 128, 64, K);
        __builtin_amdgcn_s_wait_tensorcnt(2);   // pair for slice ks complete
      } else {
        __builtin_amdgcn_s_wait_tensorcnt(0);
      }
    }
    __syncthreads();                            // publish buffer `cur`

#pragma unroll
    for (int kk = 0; kk < 2; ++kk) {            // two 32-wide WMMA k-steps
      // A fragments (16x32, ISA layout: K = {half*8+j, 16+half*8+j})
      BF16Frag a[4];
#pragma unroll
      for (int mt = 0; mt < 4; ++mt) {
        const __bf16* p = &As[cur][(wm * 64 + mt * 16 + n16) * LROW + kk * 32];
        a[mt].h[0] = *(const v8bf*)(p + half * 8);
        a[mt].h[1] = *(const v8bf*)(p + 16 + half * 8);
      }
      // B fragments (32x16, lane covers K = {half*8+j, 16+half*8+j} of column n16)
      BF16Frag b[2];
#pragma unroll
      for (int nt = 0; nt < 2; ++nt) {
        const __bf16* p = &Bs[cur][(wn * 32 + nt * 16 + n16) * LROW + kk * 32 + half * 8];
        b[nt].h[0] = *(const v8bf*)(p);
        b[nt].h[1] = *(const v8bf*)(p + 16);
      }
#pragma unroll
      for (int mt = 0; mt < 4; ++mt)
#pragma unroll
        for (int nt = 0; nt < 2; ++nt)
          acc[mt][nt].v = __builtin_amdgcn_wmma_f32_16x16x32_bf16(
              false, a[mt].v, false, b[nt].v, (short)0, acc[mt][nt].v, false, false);
    }
    __syncthreads();                            // protect `cur` before reuse
  }

  // ---- epilogue (D layout: lane n16 = column, row m = half*8 + r)
#pragma unroll
  for (int mt = 0; mt < 4; ++mt) {
#pragma unroll
    for (int nt = 0; nt < 2; ++nt) {
      const int n_g = n0 + wn * 32 + nt * 16 + n16;
      const float bv = bias[n_g];
      if (MODE == 0) {
#pragma unroll
        for (int r = 0; r < 8; ++r) {
          const int m_g = m0 + wm * 64 + mt * 16 + half * 8 + r;
          outp[(size_t)m_g * N + n_g] = acc[mt][nt].f[r] + bv;
        }
      } else {
        const int which = n_g >> 10;          // 0=q 1=k 2=v
        const int cc = n_g & 1023;
        const int hh = cc >> 6;
        const int dd = cc & 63;
#pragma unroll
        for (int r = 0; r < 8; ++r) {
          const int m_g = m0 + wm * 64 + mt * 16 + half * 8 + r;
          const int bb = m_g >> 11;           // batch
          const int tt = m_g & 2047;          // time
          const __bf16 val = f2bf(acc[mt][nt].f[r] + bv);
          if (which == 2) {
            Vt[((size_t)(bb * H_SZ + hh) * D_SZ + dd) * T_SZ + tt] = val;
          } else {
            const size_t idx = ((size_t)(bb * H_SZ + hh) * T_SZ + tt) * D_SZ + dd;
            (which == 0 ? Qb : Kb)[idx] = val;
          }
        }
      }
    }
  }
}

// ---------------------------------------------------------------------------
// 4) flash attention: block = 8 waves, wave owns 16 query rows (128 rows/WG).
//    Per 32-key step: 4 WMMA for S = Q@K^T, online softmax (half-wave
//    shuffle reductions), P via wave-private LDS, 4 WMMA for O += P@V.
//    No block barriers (waves have divergent causal trip counts).
// ---------------------------------------------------------------------------
__global__ __launch_bounds__(256) void flash_attn(
    const __bf16* __restrict__ Q, const __bf16* __restrict__ Kt,
    const __bf16* __restrict__ Vt, __bf16* __restrict__ O) {
  __shared__ __attribute__((aligned(16))) __bf16 Pl[8 * 16 * 40];

  const int tid  = threadIdx.x;
  const int lane = tid & 31;
  const int w    = tid >> 5;
  const int n16  = lane & 15;
  const int half = lane >> 4;

  const int bh = blockIdx.y;               // 0..31
  const int bb = bh >> 4;
  const int hh = bh & 15;
  const int q0 = blockIdx.x * 128;
  const int qbase = q0 + w * 16;           // this wave's first query row
  const int q_hi  = qbase + 15;

  __bf16* Pw = Pl + w * (16 * 40);         // wave-private P tile [16][40]

  // Q fragments (16 x 64 split into two 16x32 A-fragments)
  const __bf16* qp = Q + ((size_t)bh * T_SZ + qbase + n16) * D_SZ;
  BF16Frag aQ[2];
#pragma unroll
  for (int kk = 0; kk < 2; ++kk) {
    aQ[kk].h[0] = *(const v8bf*)(qp + kk * 32 + half * 8);
    aQ[kk].h[1] = *(const v8bf*)(qp + kk * 32 + 16 + half * 8);
  }

  const float NEG = -1.0e30f;
  float mrun[8], lrun[8];
  FAcc o[4];
#pragma unroll
  for (int r = 0; r < 8; ++r) { mrun[r] = NEG; lrun[r] = 0.0f; }
#pragma unroll
  for (int dt = 0; dt < 4; ++dt)
#pragma unroll
    for (int r = 0; r < 8; ++r) o[dt].f[r] = 0.0f;

  const __bf16* Kg = Kt + (size_t)bh * T_SZ * D_SZ;
  const __bf16* Vg = Vt + (size_t)bh * D_SZ * T_SZ;

  for (int kt0 = 0; kt0 <= q_hi; kt0 += 32) {
    // ---- S = Q @ K^T  (two 16x16 column tiles)
    FAcc s[2];
#pragma unroll
    for (int st = 0; st < 2; ++st) {
#pragma unroll
      for (int r = 0; r < 8; ++r) s[st].f[r] = 0.0f;
#pragma unroll
      for (int kk = 0; kk < 2; ++kk) {
        const __bf16* kp = Kg + (size_t)(kt0 + st * 16 + n16) * D_SZ + kk * 32 + half * 8;
        BF16Frag bK;
        bK.h[0] = *(const v8bf*)(kp);
        bK.h[1] = *(const v8bf*)(kp + 16);
        s[st].v = __builtin_amdgcn_wmma_f32_16x16x32_bf16(
            false, aQ[kk].v, false, bK.v, (short)0, s[st].v, false, false);
      }
    }

    // ---- scale + causal mask (D layout: row = half*8+r, col = kt0+st*16+n16)
#pragma unroll
    for (int st = 0; st < 2; ++st)
#pragma unroll
      for (int r = 0; r < 8; ++r) {
        const int key = kt0 + st * 16 + n16;
        const int qg  = qbase + half * 8 + r;
        const float sv = s[st].f[r] * 0.125f;       // 1/sqrt(64)
        s[st].f[r] = (key <= qg) ? sv : NEG;
      }

    // ---- online softmax (rows live across the 16 lanes of each half-wave)
    float alpha[8];
#pragma unroll
    for (int r = 0; r < 8; ++r) {
      float rm = fmaxf(s[0].f[r], s[1].f[r]);
      rm = fmaxf(rm, __shfl_xor(rm, 1, 32));
      rm = fmaxf(rm, __shfl_xor(rm, 2, 32));
      rm = fmaxf(rm, __shfl_xor(rm, 4, 32));
      rm = fmaxf(rm, __shfl_xor(rm, 8, 32));
      const float mn = fmaxf(mrun[r], rm);
      const float al = __expf(mrun[r] - mn);
      const float p0 = __expf(s[0].f[r] - mn);
      const float p1 = __expf(s[1].f[r] - mn);
      s[0].f[r] = p0; s[1].f[r] = p1;
      float rs = p0 + p1;
      rs += __shfl_xor(rs, 1, 32);
      rs += __shfl_xor(rs, 2, 32);
      rs += __shfl_xor(rs, 4, 32);
      rs += __shfl_xor(rs, 8, 32);
      lrun[r] = lrun[r] * al + rs;
      mrun[r] = mn;
      alpha[r] = al;
    }
#pragma unroll
    for (int dt = 0; dt < 4; ++dt)
#pragma unroll
      for (int r = 0; r < 8; ++r) o[dt].f[r] *= alpha[r];

    // ---- P (16x32 bf16) through wave-private LDS: D-layout -> A-fragment
#pragma unroll
    for (int st = 0; st < 2; ++st)
#pragma unroll
      for (int r = 0; r < 8; ++r)
        Pw[(half * 8 + r) * 40 + st * 16 + n16] = f2bf(s[st].f[r]);

    // wave-private LDS RAW: DS ops are in-order per wave; force the wait.
    asm volatile("s_wait_dscnt 0" ::: "memory");

    BF16Frag aP;
    aP.h[0] = *(const v8bf*)(Pw + n16 * 40 + half * 8);
    aP.h[1] = *(const v8bf*)(Pw + n16 * 40 + 16 + half * 8);

    // ---- O += P @ V   (V stored [d][t], so B-fragment is contiguous)
#pragma unroll
    for (int dt = 0; dt < 4; ++dt) {
      const __bf16* vp = Vg + (size_t)(dt * 16 + n16) * T_SZ + kt0 + half * 8;
      BF16Frag bV;
      bV.h[0] = *(const v8bf*)(vp);
      bV.h[1] = *(const v8bf*)(vp + 16);
      o[dt].v = __builtin_amdgcn_wmma_f32_16x16x32_bf16(
          false, aP.v, false, bV.v, (short)0, o[dt].v, false, false);
    }
  }

  // ---- normalize + store attention output as bf16 [B,T,C], c = h*64+d
  float rinv[8];
#pragma unroll
  for (int r = 0; r < 8; ++r) rinv[r] = 1.0f / lrun[r];
#pragma unroll
  for (int dt = 0; dt < 4; ++dt)
#pragma unroll
    for (int r = 0; r < 8; ++r) {
      const int tg = qbase + half * 8 + r;
      const int c  = hh * D_SZ + dt * 16 + n16;
      O[((size_t)bb * T_SZ + tg) * C_SZ + c] = f2bf(o[dt].f[r] * rinv[r]);
    }
}

// ---------------------------------------------------------------------------
// launch
// ---------------------------------------------------------------------------
extern "C" void kernel_launch(void* const* d_in, const int* in_sizes, int n_in,
                              void* d_out, int out_size, void* d_ws, size_t ws_size,
                              hipStream_t stream) {
  (void)in_sizes; (void)n_in; (void)out_size; (void)ws_size;
  const float* x    = (const float*)d_in[0];
  const float* Wqkv = (const float*)d_in[1];
  const float* bqkv = (const float*)d_in[2];
  const float* Wout = (const float*)d_in[3];
  const float* bout = (const float*)d_in[4];
  float* out = (float*)d_out;

  char* ws = (char*)d_ws;
  __bf16* xb    = (__bf16*)(ws);                          //  8 MiB [4096,1024]
  __bf16* WqkvT = (__bf16*)(ws + ((size_t)8  << 20));     //  6 MiB [3072,1024]
  __bf16* WoutT = (__bf16*)(ws + ((size_t)14 << 20));     //  2 MiB [1024,1024]
  __bf16* Qb    = (__bf16*)(ws + ((size_t)16 << 20));     //  8 MiB [B,H,T,D]
  __bf16* Kb    = (__bf16*)(ws + ((size_t)24 << 20));     //  8 MiB [B,H,T,D]
  __bf16* Vtb   = (__bf16*)(ws + ((size_t)32 << 20));     //  8 MiB [B,H,D,T]
  __bf16* Ab    = (__bf16*)(ws + ((size_t)40 << 20));     //  8 MiB [4096,1024]

  const int Mrows = B_SZ * T_SZ;                          // 4096

  cvt_f32_bf16<<<(Mrows * C_SZ) / (256 * 4), 256, 0, stream>>>(x, xb, Mrows * C_SZ);
  transpose_f32_bf16<<<dim3(3 * C_SZ / 32, C_SZ / 32), dim3(32, 8), 0, stream>>>(
      Wqkv, WqkvT, C_SZ, 3 * C_SZ);
  transpose_f32_bf16<<<dim3(C_SZ / 32, C_SZ / 32), dim3(32, 8), 0, stream>>>(
      Wout, WoutT, C_SZ, C_SZ);

  gemm_bf16<1><<<dim3(3 * C_SZ / 128, Mrows / 128), 256, 0, stream>>>(
      xb, WqkvT, bqkv, nullptr, Qb, Kb, Vtb, Mrows, 3 * C_SZ, C_SZ);

  flash_attn<<<dim3(T_SZ / 128, B_SZ * H_SZ), 256, 0, stream>>>(Qb, Kb, Vtb, Ab);

  gemm_bf16<0><<<dim3(C_SZ / 128, Mrows / 128), 256, 0, stream>>>(
      Ab, WoutT, bout, out, nullptr, nullptr, nullptr, Mrows, C_SZ, C_SZ);
}
